// MultiHeadAttention_9775345565972
// MI455X (gfx1250) — matrix operand order repaired and checked by reading, not compiled, not run on hardware
//
#include <hip/hip_runtime.h>
#include <hip/hip_bf16.h>
#include <stdint.h>

// Shapes (fixed by the reference)
#define BB 4
#define LL 2048
#define CC 1024
#define HH 16
#define HD 64
#define QKV_STRIDE (3 * CC)   // 3072

typedef __attribute__((ext_vector_type(16))) __bf16 v16bf;
typedef __attribute__((ext_vector_type(8)))  float  v8f;
typedef __attribute__((ext_vector_type(4)))  __bf16 v4bf;
typedef __attribute__((ext_vector_type(4)))  unsigned int v4u;
typedef __attribute__((ext_vector_type(8)))  int v8i;
typedef __attribute__((ext_vector_type(4)))  int v4i;

union BF16x16 {
    v16bf    v;
    uint32_t u[8];
    uint4    q[2];
};

// ---------------------------------------------------------------------------
// TDM: 2-D tile load Global -> LDS via Tensor Data Mover (6-arg builtin form:
// groups 0/1 + two 4-dword groups + trailing 8-dword group + cache policy).
// Descriptor per CDNA5 ISA ch.8: group0 = {ctrl, lds_addr, global_addr, type},
// group1 = {mask/data_size, tensor dims, tile dims, dim0 stride}.
// ---------------------------------------------------------------------------
static __device__ __forceinline__ void tdm_load_tile_2d(
    __bf16* lds_dst, const __bf16* gsrc,
    int tile_rows, int tile_cols,
    int tensor_rows, int tensor_cols, int row_stride_elems)
{
    const uint64_t ga  = (uint64_t)(uintptr_t)gsrc;
    const uint32_t lds = (uint32_t)(uintptr_t)lds_dst;   // LDS aperture: low 32 = offset

    v4u g0;
    g0.x = 1u;                                            // count=1, user mode
    g0.y = lds;                                           // lds_addr (bytes)
    g0.z = (uint32_t)ga;                                  // global_addr[31:0]
    g0.w = (uint32_t)((ga >> 32) & 0x01FFFFFFu) | (2u << 30);  // addr[56:32], type=2

    v8i g1;
    g1[0] = (int)(1u << 16);                              // data_size=1 (2 bytes), mask=0
    g1[1] = (int)(((uint32_t)tensor_cols & 0xFFFFu) << 16);          // tensor_dim0[15:0]
    g1[2] = (int)((((uint32_t)tensor_cols) >> 16) |
                  (((uint32_t)tensor_rows & 0xFFFFu) << 16));        // dim0[31:16] | dim1[15:0]
    g1[3] = (int)((((uint32_t)tensor_rows) >> 16) |
                  (((uint32_t)tile_cols) << 16));                    // dim1[31:16] | tile_dim0
    g1[4] = tile_rows;                                    // tile_dim1 (tile_dim2 = 0)
    g1[5] = row_stride_elems;                             // tensor_dim0_stride[31:0]
    g1[6] = 0;
    g1[7] = 0;

    v4i gz4 = {0, 0, 0, 0};
    v8i gz8 = {0, 0, 0, 0, 0, 0, 0, 0};
    __builtin_amdgcn_tensor_load_to_lds(g0, g1, gz4, gz4, gz8, 0);
}

// ---------------------------------------------------------------------------
// Kernel 0: fp32 -> bf16 convert (one shot; keeps conversions out of GEMMs)
// ---------------------------------------------------------------------------
__global__ void __launch_bounds__(256)
cvt_f32_bf16_kernel(const float* __restrict__ in, __bf16* __restrict__ out, int n4)
{
    const int i = blockIdx.x * blockDim.x + threadIdx.x;
    if (i < n4) {
        float4 f = ((const float4*)in)[i];
        v4bf o;
        o.x = (__bf16)f.x; o.y = (__bf16)f.y; o.z = (__bf16)f.z; o.w = (__bf16)f.w;
        ((v4bf*)out)[i] = o;
    }
}

// ---------------------------------------------------------------------------
// Shared GEMM body: C[m,n] = sum_k A[m,k]*W[n,k] (+bias).  K = 1024.
// Block: 256 thr = 8 waves arranged 4(M) x 2(N); block tile 64M x 128N.
// A tile staged cooperatively (b128), W tile staged by the TDM.
// ---------------------------------------------------------------------------
#define GEMM_BODY(ABASE, WBASE, KDIM, NROWS_W, EPILOG)                          \
    const int lane = threadIdx.x & 31;                                          \
    const int wv   = threadIdx.x >> 5;                                          \
    const int r    = lane & 15;                                                 \
    const int hh   = lane >> 4;                                                 \
    const int mi   = wv & 3;                                                    \
    const int ni   = wv >> 2;                                                   \
    const int mblock = blockIdx.x * 64;                                         \
    const int nblock = blockIdx.y * 128;                                        \
    __shared__ __align__(16) __bf16 Abuf[64][32];                               \
    __shared__ __align__(16) __bf16 Bbuf[128][32];                              \
    v8f zf = {0.f,0.f,0.f,0.f,0.f,0.f,0.f,0.f};                                 \
    v8f acc[4] = {zf, zf, zf, zf};                                              \
    const int arow = threadIdx.x >> 2, acol = (threadIdx.x & 3) * 8;            \
    const __bf16* aptr = (ABASE) + (size_t)(mblock + arow) * (KDIM) + acol;     \
    for (int k0 = 0; k0 < (KDIM); k0 += 32) {                                   \
        __syncthreads();                                                        \
        *(uint4*)&Abuf[arow][acol] = *(const uint4*)(aptr + k0);                \
        if (threadIdx.x == 0) {                                                 \
            tdm_load_tile_2d(&Bbuf[0][0],                                       \
                             (WBASE) + (size_t)nblock * (KDIM) + k0,            \
                             128, 32, (NROWS_W), (KDIM), (KDIM));               \
            __builtin_amdgcn_s_wait_tensorcnt(0);                               \
        }                                                                       \
        if (k0 + 32 < (KDIM)) __builtin_prefetch(aptr + k0 + 32, 0, 3);         \
        __syncthreads();                                                        \
        BF16x16 a;                                                              \
        _Pragma("unroll")                                                       \
        for (int j = 0; j < 8; ++j) {                                           \
            const int kk = 16 * hh + 2 * j;                                     \
            a.u[j] = *(const uint32_t*)&Abuf[mi * 16 + r][kk];                  \
        }                                                                       \
        _Pragma("unroll")                                                       \
        for (int t = 0; t < 4; ++t) {                                           \
            const __bf16* bp = &Bbuf[ni * 64 + t * 16 + r][hh * 16];            \
            BF16x16 bf;                                                         \
            bf.q[0] = *(const uint4*)(bp);                                      \
            bf.q[1] = *(const uint4*)(bp + 8);                                  \
            acc[t] = __builtin_amdgcn_wmma_f32_16x16x32_bf16(                   \
                false, a.v, false, bf.v, (short)0, acc[t], false, false);       \
        }                                                                       \
    }                                                                           \
    EPILOG

// Kernel 1: QKV projection, bf16 output into interleaved [B,L,H,3*HD] layout.
__global__ void __launch_bounds__(256)
qkv_proj_kernel(const __bf16* __restrict__ xbf, const __bf16* __restrict__ wbf,
                const float* __restrict__ bias, __bf16* __restrict__ out)
{
    GEMM_BODY(xbf, wbf, CC, QKV_STRIDE,
    {
        _Pragma("unroll")
        for (int t = 0; t < 4; ++t) {
            const int col = nblock + ni * 64 + t * 16 + r;
            const float bv = bias[col];
            _Pragma("unroll")
            for (int j = 0; j < 8; ++j) {
                const int row = mblock + mi * 16 + j + 8 * hh;
                out[(size_t)row * QKV_STRIDE + col] = (__bf16)(acc[t][j] + bv);
            }
        }
    })
}

// Kernel 3: output projection, fp32 result + bias into d_out.
__global__ void __launch_bounds__(256)
out_proj_kernel(const __bf16* __restrict__ ctx, const __bf16* __restrict__ wbf,
                const float* __restrict__ bias, float* __restrict__ out)
{
    GEMM_BODY(ctx, wbf, CC, CC,
    {
        _Pragma("unroll")
        for (int t = 0; t < 4; ++t) {
            const int col = nblock + ni * 64 + t * 16 + r;
            const float bv = bias[col];
            _Pragma("unroll")
            for (int j = 0; j < 8; ++j) {
                const int row = mblock + mi * 16 + j + 8 * hh;
                out[(size_t)row * CC + col] = acc[t][j] + bv;
            }
        }
    })
}

// ---------------------------------------------------------------------------
// Kernel 2: flash-style attention over bf16 qkv (interleaved per-head layout).
// Grid: (B*H, L/64).  Block: 128 threads = 4 waves; wave handles 16 q-rows.
// ---------------------------------------------------------------------------
__global__ void __launch_bounds__(128)
attn_kernel(const __bf16* __restrict__ qkv, __bf16* __restrict__ ctx)
{
    const int bh   = blockIdx.x;
    const int b    = bh >> 4;
    const int h    = bh & 15;
    const int lane = threadIdx.x & 31;
    const int wv   = threadIdx.x >> 5;
    const int r    = lane & 15;
    const int hh   = lane >> 4;
    const int qb   = blockIdx.y * 64 + wv * 16;

    const __bf16* base = qkv + (size_t)b * LL * QKV_STRIDE + h * (3 * HD);
    // row l of this (b,h): base + l*QKV_STRIDE;  q at +0, k at +64, v at +128

    __shared__ __align__(16) __bf16 vT[HD][64];        // v transposed: [d][key]
    __shared__ __align__(16) __bf16 Pls[4][16][64];    // per-wave P scratch

    // ---- q fragments (2 K-steps of 32 over head dim 64), resident in VGPRs
    BF16x16 qf[2];
#pragma unroll
    for (int t = 0; t < 2; ++t) {
#pragma unroll
        for (int j = 0; j < 8; ++j) {
            const int dd = t * 32 + 16 * hh + 2 * j;
            qf[t].u[j] = *(const uint32_t*)(base + (size_t)(qb + r) * QKV_STRIDE + dd);
        }
    }

    v8f zacc = {0.f,0.f,0.f,0.f,0.f,0.f,0.f,0.f};
    v8f accO[4] = {zacc, zacc, zacc, zacc};
    float mrow[8], lrow[8];
#pragma unroll
    for (int j = 0; j < 8; ++j) { mrow[j] = -1e30f; lrow[j] = 0.f; }

    const float scale = 0.125f;   // 1/sqrt(64)

    for (int kb = 0; kb < LL; kb += 64) {
        __syncthreads();
        // ---- cooperative load of V block, stored transposed into LDS
#pragma unroll
        for (int i = 0; i < 16; ++i) {
            const int p   = threadIdx.x + i * 128;   // 0..2047 (64 keys * 32 pairs)
            const int key = p >> 5;
            const int dp  = p & 31;
            const uint32_t u =
                *(const uint32_t*)(base + (size_t)(kb + key) * QKV_STRIDE + 128 + 2 * dp);
            *(uint16_t*)&vT[2 * dp][key]     = (uint16_t)(u & 0xFFFFu);
            *(uint16_t*)&vT[2 * dp + 1][key] = (uint16_t)(u >> 16);
        }
        __syncthreads();

        // ---- S = q . k^T  (16 x 64): 4 key-tiles, K=64 -> 2 WMMAs each
        v8f s[4];
#pragma unroll
        for (int nt = 0; nt < 4; ++nt) {
            s[nt] = zacc;
#pragma unroll
            for (int t = 0; t < 2; ++t) {
                BF16x16 kf;
                const __bf16* kp =
                    base + (size_t)(kb + nt * 16 + r) * QKV_STRIDE + HD + t * 32 + hh * 16;
                kf.q[0] = *(const uint4*)(kp);
                kf.q[1] = *(const uint4*)(kp + 8);
                s[nt] = __builtin_amdgcn_wmma_f32_16x16x32_bf16(
                    false, qf[t].v, false, kf.v, (short)0, s[nt], false, false);
            }
#pragma unroll
            for (int j = 0; j < 8; ++j) s[nt][j] *= scale;
        }

        // ---- online softmax (row j+8*hh in slot j, striped over 16 lanes)
        float pv[4][8];
#pragma unroll
        for (int j = 0; j < 8; ++j) {
            float mx = fmaxf(fmaxf(s[0][j], s[1][j]), fmaxf(s[2][j], s[3][j]));
#pragma unroll
            for (int off = 1; off < 16; off <<= 1)
                mx = fmaxf(mx, __shfl_xor(mx, off, 32));
            const float Mn    = fmaxf(mrow[j], mx);
            const float alpha = __expf(mrow[j] - Mn);
            mrow[j] = Mn;
            float sum = 0.f;
#pragma unroll
            for (int nt = 0; nt < 4; ++nt) {
                pv[nt][j] = __expf(s[nt][j] - Mn);
                sum += pv[nt][j];
            }
#pragma unroll
            for (int off = 1; off < 16; off <<= 1)
                sum += __shfl_xor(sum, off, 32);
            lrow[j] = lrow[j] * alpha + sum;
#pragma unroll
            for (int nt = 0; nt < 4; ++nt) accO[nt][j] *= alpha;
#pragma unroll
            for (int nt = 0; nt < 4; ++nt)
                Pls[wv][j + 8 * hh][nt * 16 + r] = (__bf16)pv[nt][j];
        }

        // ---- reload P as A-fragments (wave-private LDS; DS ops in-order)
        BF16x16 pf[2];
#pragma unroll
        for (int t = 0; t < 2; ++t) {
#pragma unroll
            for (int j = 0; j < 8; ++j) {
                const int kk = t * 32 + 16 * hh + 2 * j;
                pf[t].u[j] = *(const uint32_t*)&Pls[wv][r][kk];
            }
        }

        // ---- O += P . V   (B-fragments contiguous in vT rows)
#pragma unroll
        for (int nt = 0; nt < 4; ++nt) {
#pragma unroll
            for (int t = 0; t < 2; ++t) {
                BF16x16 vf;
                const __bf16* vp = &vT[nt * 16 + r][t * 32 + hh * 16];
                vf.q[0] = *(const uint4*)(vp);
                vf.q[1] = *(const uint4*)(vp + 8);
                accO[nt] = __builtin_amdgcn_wmma_f32_16x16x32_bf16(
                    false, pf[t].v, false, vf.v, (short)0, accO[nt], false, false);
            }
        }
    }

    // ---- epilogue: normalize, store ctx bf16 in [B, L, H*HD]
#pragma unroll
    for (int j = 0; j < 8; ++j) {
        const float inv = 1.0f / lrow[j];
        const int row = qb + j + 8 * hh;
#pragma unroll
        for (int nt = 0; nt < 4; ++nt) {
            ctx[(size_t)(b * LL + row) * CC + h * HD + nt * 16 + r] =
                (__bf16)(accO[nt][j] * inv);
        }
    }
}

// ---------------------------------------------------------------------------
extern "C" void kernel_launch(void* const* d_in, const int* in_sizes, int n_in,
                              void* d_out, int out_size, void* d_ws, size_t ws_size,
                              hipStream_t stream) {
    (void)in_sizes; (void)n_in; (void)out_size; (void)ws_size;
    const float* x     = (const float*)d_in[0];
    const float* W_qkv = (const float*)d_in[1];
    const float* b_qkv = (const float*)d_in[2];
    const float* W_out = (const float*)d_in[3];
    const float* b_out = (const float*)d_in[4];
    float* out = (float*)d_out;

    // workspace layout (bytes)
    char* ws = (char*)d_ws;
    __bf16* qkv     = (__bf16*)(ws);                               // 8192*3072*2 = 50.3MB
    __bf16* ctx     = (__bf16*)(ws + 50331648);                    // 8192*1024*2 = 16.8MB
    __bf16* xbf     = (__bf16*)(ws + 50331648 + 16777216);         // 16.8MB
    __bf16* wqkv_bf = (__bf16*)(ws + 50331648 + 2*16777216);       // 6.3MB
    __bf16* wout_bf = (__bf16*)(ws + 50331648 + 2*16777216 + 6291456); // 2.1MB

    // one-shot fp32 -> bf16 conversions
    cvt_f32_bf16_kernel<<<(BB*LL*CC/4 + 255)/256, 256, 0, stream>>>(x, xbf, BB*LL*CC/4);
    cvt_f32_bf16_kernel<<<(QKV_STRIDE*CC/4 + 255)/256, 256, 0, stream>>>(W_qkv, wqkv_bf, QKV_STRIDE*CC/4);
    cvt_f32_bf16_kernel<<<(CC*CC/4 + 255)/256, 256, 0, stream>>>(W_out, wout_bf, CC*CC/4);

    // M=8192 -> 128 blocks of 64 rows; N_qkv=3072 -> 24 blocks of 128 cols
    qkv_proj_kernel<<<dim3(128, 24), 256, 0, stream>>>(xbf, wqkv_bf, b_qkv, qkv);
    // (B*H)=64 blocks x, L/64=32 blocks y, 4 waves each
    attn_kernel<<<dim3(64, 32), 128, 0, stream>>>(qkv, ctx);
    // N_out=1024 -> 8 blocks of 128 cols
    out_proj_kernel<<<dim3(128, 8), 256, 0, stream>>>(ctx, wout_bf, b_out, out);
}
